// swin3plus_61701500174571
// MI455X (gfx1250) — matrix operand order repaired and checked by reading, not compiled, not run on hardware
//
#include <hip/hip_runtime.h>
#include <math.h>

// ---------------------------------------------------------------------------
// Swin3D (4 blocks) + dw-conv + pw-conv + BN + ReLU for MI455X (gfx1250).
// GEMMs: LDS-tiled wave32 WMMA (v_wmma_f32_16x16x32_f16, f32 accumulate).
//   - 64x64 block tile, 8 waves, 2 accumulators/wave (A-frag reuse)
//   - K-panels staged in LDS via b128 global loads + f16 convert
//   - fragments read from LDS as aligned 16B ds_load_b128
//   - __builtin_prefetch on next K-panel (global_prefetch_b8)
// d_ws requirement: ~202 MB (50,332,160 floats).
// ---------------------------------------------------------------------------

typedef __attribute__((ext_vector_type(16))) _Float16 v16h;
typedef __attribute__((ext_vector_type(8)))  _Float16 v8h;
typedef __attribute__((ext_vector_type(8)))  float    v8f;

#define CDIM   128
#define LTOK   32768         // 32*32*32 tokens per batch
#define NROWS  65536         // B * LTOK
#define NHEAD  8
#define SHIFTV 2

__device__ __forceinline__ _Float16 f2h(float f) { return (_Float16)f; }

// shifted-window region label helper (H=W=T=32, WS=4, SHIFT=2)
__device__ __forceinline__ int regz(int z) { return z < 28 ? 0 : (z < 30 ? 1 : 2); }

// ---------------------------------------------------------------------------
// [B,C,H,W,T] -> token-major [B*L, C]
// ---------------------------------------------------------------------------
__global__ void permute_in(const float* __restrict__ x, float* __restrict__ t) {
  size_t idx = (size_t)blockIdx.x * blockDim.x + threadIdx.x; // 8,388,608
  int c = idx & 127;
  int row = (int)(idx >> 7);
  int b = row >> 15, p = row & 32767;
  t[idx] = x[((size_t)b * CDIM + c) * LTOK + p];
}

// ---------------------------------------------------------------------------
// LayerNorm over C=128; mode=0: identity row map; mode=1: roll(-shift) +
// 4x4x4 window partition (output row = widx*64 + n).
// ---------------------------------------------------------------------------
__global__ __launch_bounds__(128)
void ln_kernel(const float* __restrict__ t, const float* __restrict__ gamma,
               const float* __restrict__ beta, float* __restrict__ out,
               int shift, int mode) {
  const int r = blockIdx.x;
  const int c = threadIdx.x;
  int srcrow;
  if (mode == 0) {
    srcrow = r;
  } else {
    int widx = r >> 6, n = r & 63;
    int b = widx >> 9, wloc = widx & 511;
    int wh = wloc >> 6, ww = (wloc >> 3) & 7, wt = wloc & 7;
    int a = n >> 4, bb = (n >> 2) & 3, cc = n & 3;
    int gh = (wh * 4 + a + shift) & 31;
    int gw = (ww * 4 + bb + shift) & 31;
    int gt = (wt * 4 + cc + shift) & 31;
    srcrow = (b << 15) + (gh << 10) + (gw << 5) + gt;
  }
  float v = t[(size_t)srcrow * CDIM + c];
  __shared__ float red[128];
  red[c] = v; __syncthreads();
  for (int s = 64; s > 0; s >>= 1) { if (c < s) red[c] += red[c + s]; __syncthreads(); }
  float mu = red[0] * (1.0f / 128.0f);
  __syncthreads();
  float d = v - mu;
  red[c] = d * d; __syncthreads();
  for (int s = 64; s > 0; s >>= 1) { if (c < s) red[c] += red[c + s]; __syncthreads(); }
  float var = red[0] * (1.0f / 128.0f);
  out[(size_t)r * CDIM + c] = d * rsqrtf(var + 1e-5f) * gamma[c] + beta[c];
}

// ---------------------------------------------------------------------------
// LDS-tiled WMMA GEMM: out[M,Nc] = act(A[M,K] @ W[Nc,K]^T + bias) (+ res)
// Grid: (Nc/64, M/64); block: 256 threads = 8 waves.
// Wave wv handles tile row mi = wv&3 and tile cols ni = (wv>>2)*2 + {0,1}.
// LDS panels: sA[64 rows][32 K] f16, sB[64 cols][32 K] f16, rows padded to 40
// halves (80 B) so 16B ds_load_b128 fragment reads are conflict-friendly.
// Fragment k order: contiguous K = g*16 + h for both A and B (16-bit A 16x32 / B 32x16).
// ---------------------------------------------------------------------------
__global__ __launch_bounds__(256)
void gemm_wmma(const float* __restrict__ A, const float* __restrict__ W,
               const float* __restrict__ bias, const float* __restrict__ res,
               float* __restrict__ out, int M, int K, int Nc, int act) {
  __shared__ _Float16 sA[64][40];
  __shared__ _Float16 sB[64][40];

  const int tid  = threadIdx.x;
  const int lane = tid & 31;
  const int wv   = tid >> 5;          // 0..7
  const int g    = lane >> 4;         // K sub-block select
  const int mr   = lane & 15;         // row (A) / col (B,D)

  const int m0base = blockIdx.y * 64;
  const int n0base = blockIdx.x * 64;
  const int mi = wv & 3;              // A tile row (16 rows)
  const int nb = (wv >> 2) * 2;       // first of two B tiles

  // cooperative loader mapping: 64 rows x 32 K, 8 floats per thread
  const int ldrow = tid >> 2;
  const int ldcol = (tid & 3) * 8;
  const float* aptr = A + (size_t)(m0base + ldrow) * K + ldcol;
  const float* wptr = W + (size_t)(n0base + ldrow) * K + ldcol;

  v8f acc0 = {}, acc1 = {};

  for (int k0 = 0; k0 < K; k0 += 32) {
    __syncthreads();   // previous iteration's LDS reads complete
    {
      const float4 a0 = *(const float4*)(aptr + k0);
      const float4 a1 = *(const float4*)(aptr + k0 + 4);
      const float4 b0 = *(const float4*)(wptr + k0);
      const float4 b1 = *(const float4*)(wptr + k0 + 4);
      if (k0 + 32 < K) {              // prefetch next K-panel
        __builtin_prefetch(aptr + k0 + 32, 0, 3);
        __builtin_prefetch(wptr + k0 + 32, 0, 3);
      }
      sA[ldrow][ldcol + 0] = f2h(a0.x); sA[ldrow][ldcol + 1] = f2h(a0.y);
      sA[ldrow][ldcol + 2] = f2h(a0.z); sA[ldrow][ldcol + 3] = f2h(a0.w);
      sA[ldrow][ldcol + 4] = f2h(a1.x); sA[ldrow][ldcol + 5] = f2h(a1.y);
      sA[ldrow][ldcol + 6] = f2h(a1.z); sA[ldrow][ldcol + 7] = f2h(a1.w);
      sB[ldrow][ldcol + 0] = f2h(b0.x); sB[ldrow][ldcol + 1] = f2h(b0.y);
      sB[ldrow][ldcol + 2] = f2h(b0.z); sB[ldrow][ldcol + 3] = f2h(b0.w);
      sB[ldrow][ldcol + 4] = f2h(b1.x); sB[ldrow][ldcol + 5] = f2h(b1.y);
      sB[ldrow][ldcol + 6] = f2h(b1.z); sB[ldrow][ldcol + 7] = f2h(b1.w);
    }
    __syncthreads();

    // A fragment (shared by both N tiles): halves 0..15 -> K=g*16+h
    // (contiguous, matching B)
    const int arow = mi * 16 + mr;
    v8h alo = *(const v8h*)&sA[arow][g * 16];
    v8h ahi = *(const v8h*)&sA[arow][g * 16 + 8];
    v16h av = __builtin_shufflevector(alo, ahi, 0, 1, 2, 3, 4, 5, 6, 7,
                                      8, 9, 10, 11, 12, 13, 14, 15);
    // B fragments: lane col = mr, K = g*16 + h (contiguous, matching A)
    const int b0row = (nb + 0) * 16 + mr;
    const int b1row = (nb + 1) * 16 + mr;
    v8h p0 = *(const v8h*)&sB[b0row][g * 16];
    v8h p1 = *(const v8h*)&sB[b0row][g * 16 + 8];
    v16h bv0 = __builtin_shufflevector(p0, p1, 0, 1, 2, 3, 4, 5, 6, 7,
                                       8, 9, 10, 11, 12, 13, 14, 15);
    v8h q0 = *(const v8h*)&sB[b1row][g * 16];
    v8h q1 = *(const v8h*)&sB[b1row][g * 16 + 8];
    v16h bv1 = __builtin_shufflevector(q0, q1, 0, 1, 2, 3, 4, 5, 6, 7,
                                       8, 9, 10, 11, 12, 13, 14, 15);

    acc0 = __builtin_amdgcn_wmma_f32_16x16x32_f16(false, av, false, bv0,
                                                  (short)0, acc0, false, false);
    acc1 = __builtin_amdgcn_wmma_f32_16x16x32_f16(false, av, false, bv1,
                                                  (short)0, acc1, false, false);
  }

#pragma unroll
  for (int half = 0; half < 2; ++half) {
    const v8f acc = half ? acc1 : acc0;
    const int col = n0base + (nb + half) * 16 + mr;
#pragma unroll
    for (int r = 0; r < 8; ++r) {
      int row = m0base + mi * 16 + r + g * 8;
      float v = acc[r] + bias[col];
      if (act == 1) v = 0.5f * v * (1.0f + erff(v * 0.70710678118654752f));
      if (res) v += res[(size_t)row * Nc + col];
      out[(size_t)row * Nc + col] = v;
    }
  }
}

// ---------------------------------------------------------------------------
// Windowed attention: one block (128 thr = 4 waves) per (window, head).
// S = (Q*scale) K^T via WMMA (hd=16 zero-padded to K=32), + rel-pos bias
// + shift mask, softmax (1/sum folded into output), O = P V via 2 WMMA
// K-chunks. Output in window order [B*nW*64, 128] at cols head*16..+16.
// ---------------------------------------------------------------------------
__global__ __launch_bounds__(128)
void attn_win(const float* __restrict__ qkv, const float* __restrict__ rpb,
              float* __restrict__ owin, int shift) {
  const int w    = blockIdx.x >> 3;   // 0..1023
  const int head = blockIdx.x & 7;
  const int tid  = threadIdx.x;
  const int lane = tid & 31;
  const int wv   = tid >> 5;          // 0..3 (row strip)
  const int g    = lane >> 4;
  const int mr   = lane & 15;

  __shared__ float sS[64][65];
  __shared__ float rowScale[64];

  const float scale = 0.25f;          // 16^-0.5
  const size_t base = (size_t)w * 64 * 384;
  const int m0 = wv * 16;

  // --- Q fragment (real K in 0..15, pad 16..31 with zero) ---
  v16h qa;
  {
    const float* qrow = qkv + base + (size_t)(m0 + mr) * 384 + head * 16;
#pragma unroll
    for (int h = 0; h < 8; ++h) qa[h] = f2h(g == 0 ? qrow[h] * scale : 0.0f); // K=g*16+h, g=1 pad
#pragma unroll
    for (int h = 8; h < 16; ++h) qa[h] = f2h(g == 0 ? qrow[h] * scale : 0.0f); // K=g*16+h, g=1 pad
  }
  // --- S = Q K^T, four 16-col tiles ---
  for (int c0 = 0; c0 < 64; c0 += 16) {
    v16h kb;
    if (g == 0) { // K-dim block 0..15 (real)
      const float* krow = qkv + base + (size_t)(c0 + mr) * 384 + 128 + head * 16;
#pragma unroll
      for (int h = 0; h < 16; ++h) kb[h] = f2h(krow[h]);
    } else {      // K-dim block 16..31 (pad)
#pragma unroll
      for (int h = 0; h < 16; ++h) kb[h] = f2h(0.0f);
    }
    v8f s = {};
    s = __builtin_amdgcn_wmma_f32_16x16x32_f16(false, qa, false, kb,
                                               (short)0, s, false, false);
#pragma unroll
    for (int r = 0; r < 8; ++r) sS[m0 + r + g * 8][c0 + mr] = s[r];
  }
  __syncthreads();

  // --- bias + mask + softmax (rows handled by threads 0..63) ---
  if (tid < 64) {
    const int n = tid;
    const int i1 = n >> 4, j1 = (n >> 2) & 3, k1 = n & 3;
    const int wb = w & 511;
    const int wh = wb >> 6, ww = (wb >> 3) & 7, wt = wb & 7;
    int labn = 0;
    if (shift) {
      labn = regz(wh * 4 + i1) * 9 + regz(ww * 4 + j1) * 3 + regz(wt * 4 + k1);
    }
    float mx = -1e30f;
    for (int m = 0; m < 64; ++m) {
      int i2 = m >> 4, j2 = (m >> 2) & 3, k2 = m & 3;
      int ridx = (i1 - i2 + 3) * 49 + (j1 - j2 + 3) * 7 + (k1 - k2 + 3);
      float v = sS[n][m] + rpb[ridx * NHEAD + head];
      if (shift) {
        int labm = regz(wh * 4 + i2) * 9 + regz(ww * 4 + j2) * 3 + regz(wt * 4 + k2);
        if (labm != labn) v -= 100.0f;
      }
      sS[n][m] = v;
      mx = fmaxf(mx, v);
    }
    float sum = 0.0f;
    for (int m = 0; m < 64; ++m) {
      float e = __expf(sS[n][m] - mx);
      sS[n][m] = e;
      sum += e;
    }
    rowScale[n] = 1.0f / sum;
  }
  __syncthreads();

  // --- O = P V (K = 64, two WMMA chunks); fold 1/sum at store ---
  v8f o = {};
  for (int kc = 0; kc < 64; kc += 32) {
    v16h pa, vb;
#pragma unroll
    for (int h = 0; h < 8; ++h) pa[h] = f2h(sS[m0 + mr][kc + h + g * 8]);
#pragma unroll
    for (int h = 0; h < 8; ++h) pa[h + 8] = f2h(sS[m0 + mr][kc + 16 + h + g * 8]);
    const int kbb = g * 8;
#pragma unroll
    for (int h = 0; h < 16; ++h) {
      int ktok = kc + kbb + (h < 8 ? h : h + 8);
      vb[h] = f2h(qkv[base + (size_t)ktok * 384 + 256 + head * 16 + mr]);
    }
    o = __builtin_amdgcn_wmma_f32_16x16x32_f16(false, pa, false, vb,
                                               (short)0, o, false, false);
  }
#pragma unroll
  for (int r = 0; r < 8; ++r) {
    int row = m0 + r + g * 8;
    owin[((size_t)w * 64 + row) * CDIM + head * 16 + mr] = o[r] * rowScale[row];
  }
}

// ---------------------------------------------------------------------------
// window reverse + roll(+shift) + residual add into t
// ---------------------------------------------------------------------------
__global__ void reverse_add(float* __restrict__ t, const float* __restrict__ pout,
                            int shift) {
  size_t idx = (size_t)blockIdx.x * blockDim.x + threadIdx.x; // 8,388,608
  int c = idx & 127;
  int row = (int)(idx >> 7);
  int b = row >> 15, p = row & 32767;
  int ph = p >> 10, pw = (p >> 5) & 31, pt = p & 31;
  int qh = (ph - shift + 32) & 31, qw = (pw - shift + 32) & 31, qt = (pt - shift + 32) & 31;
  int wh = qh >> 2, a = qh & 3, ww = qw >> 2, bb = qw & 3, wt = qt >> 2, cc = qt & 3;
  int widx = (b << 9) + (wh << 6) + (ww << 3) + wt;
  int n = (a << 4) + (bb << 2) + cc;
  t[idx] += pout[((size_t)widx * 64 + n) * CDIM + c];
}

// ---------------------------------------------------------------------------
// depthwise 3x3x3 conv (SAME, zero pad), token-major in/out
// ---------------------------------------------------------------------------
__global__ void dwconv(const float* __restrict__ t, const float* __restrict__ w,
                       const float* __restrict__ bias, float* __restrict__ out) {
  size_t idx = (size_t)blockIdx.x * blockDim.x + threadIdx.x; // 8,388,608
  int c = idx & 127;
  int row = (int)(idx >> 7);
  int b = row >> 15, p = row & 32767;
  int ph = p >> 10, pw = (p >> 5) & 31, pt = p & 31;
  float acc = bias[c];
#pragma unroll
  for (int dz = -1; dz <= 1; ++dz)
#pragma unroll
    for (int dy = -1; dy <= 1; ++dy)
#pragma unroll
      for (int dx = -1; dx <= 1; ++dx) {
        int zh = ph + dz, zw = pw + dy, zt = pt + dx;
        if ((unsigned)zh < 32u && (unsigned)zw < 32u && (unsigned)zt < 32u) {
          acc += t[(((size_t)(b << 15) + (zh << 10) + (zw << 5) + zt)) * CDIM + c]
               * w[c * 27 + (dz + 1) * 9 + (dy + 1) * 3 + (dx + 1)];
        }
      }
  out[idx] = acc;
}

// ---------------------------------------------------------------------------
// BatchNorm statistics over [65536, 256] (per channel), then apply + ReLU
// with permute to [B, OUT, 32^3].
// ---------------------------------------------------------------------------
__global__ __launch_bounds__(256)
void bn_stats(const float* __restrict__ z, float* __restrict__ stats) {
  int o = blockIdx.x, tid = threadIdx.x;
  float s = 0.0f, s2 = 0.0f;
  for (int r = tid; r < NROWS; r += 256) {
    float v = z[(size_t)r * 256 + o];
    s += v; s2 += v * v;
  }
  __shared__ float rs[256], rq[256];
  rs[tid] = s; rq[tid] = s2; __syncthreads();
  for (int st = 128; st > 0; st >>= 1) {
    if (tid < st) { rs[tid] += rs[tid + st]; rq[tid] += rq[tid + st]; }
    __syncthreads();
  }
  if (tid == 0) {
    float mu = rs[0] * (1.0f / (float)NROWS);
    stats[o] = mu;
    stats[256 + o] = rq[0] * (1.0f / (float)NROWS) - mu * mu;
  }
}

__global__ void bn_apply(const float* __restrict__ z, const float* __restrict__ stats,
                         const float* __restrict__ g, const float* __restrict__ b,
                         float* __restrict__ out) {
  size_t idx = (size_t)blockIdx.x * blockDim.x + threadIdx.x; // 16,777,216
  int p = (int)(idx & 32767);
  int o = (int)((idx >> 15) & 255);
  int bi = (int)(idx >> 23);
  float v = z[(((size_t)bi << 15) + p) * 256 + o];
  v = (v - stats[o]) * rsqrtf(stats[256 + o] + 1e-5f) * g[o] + b[o];
  out[idx] = fmaxf(v, 0.0f);
}

// ---------------------------------------------------------------------------
// Orchestration
// ---------------------------------------------------------------------------
extern "C" void kernel_launch(void* const* d_in, const int* in_sizes, int n_in,
                              void* d_out, int out_size, void* d_ws, size_t ws_size,
                              hipStream_t stream) {
  (void)in_sizes; (void)n_in; (void)out_size; (void)ws_size;
  const float* x       = (const float*)d_in[0];
  const float* norm1_g = (const float*)d_in[1];
  const float* norm1_b = (const float*)d_in[2];
  const float* qkv_w   = (const float*)d_in[3];
  const float* qkv_b   = (const float*)d_in[4];
  const float* proj_w  = (const float*)d_in[5];
  const float* proj_b  = (const float*)d_in[6];
  const float* rpb     = (const float*)d_in[7];
  const float* norm2_g = (const float*)d_in[8];
  const float* norm2_b = (const float*)d_in[9];
  const float* fc1_w   = (const float*)d_in[10];
  const float* fc1_b   = (const float*)d_in[11];
  const float* fc2_w   = (const float*)d_in[12];
  const float* fc2_b   = (const float*)d_in[13];
  const float* dw_w    = (const float*)d_in[14];
  const float* dw_b    = (const float*)d_in[15];
  const float* pw_w    = (const float*)d_in[16];
  const float* pw_b    = (const float*)d_in[17];
  const float* bn_g    = (const float*)d_in[18];
  const float* bn_b    = (const float*)d_in[19];

  float* ws   = (float*)d_ws;
  float* t    = ws;                       //  8,388,608 f : token buffer
  float* buf1 = ws + 8388608;             //  8,388,608 f : win / xn / proj-out / dw-out
  float* big  = ws + 16777216;            // 33,554,432 f : qkv+owin | mlp | z (aliased)
  float* qkvb = big;                      // 25,165,824 f
  float* owin = big + 25165824;           //  8,388,608 f
  float* mbuf = big;                      // 33,554,432 f (qkv/owin dead by then)
  float* zbuf = big;                      // 16,777,216 f (mlp dead by then)
  float* stats = ws + 50331648;           //        512 f

  permute_in<<<32768, 256, 0, stream>>>(x, t);

  const int shifts[4] = {0, SHIFTV, 0, SHIFTV};
  for (int i = 0; i < 4; ++i) {
    const float* n1g = norm1_g + i * CDIM;
    const float* n1b = norm1_b + i * CDIM;
    const float* qw  = qkv_w + (size_t)i * 384 * CDIM;
    const float* qb  = qkv_b + (size_t)i * 384;
    const float* pwm = proj_w + (size_t)i * CDIM * CDIM;
    const float* pbm = proj_b + (size_t)i * CDIM;
    const float* rp  = rpb + (size_t)i * 343 * NHEAD;
    const float* n2g = norm2_g + i * CDIM;
    const float* n2b = norm2_b + i * CDIM;
    const float* f1w = fc1_w + (size_t)i * 512 * CDIM;
    const float* f1b = fc1_b + (size_t)i * 512;
    const float* f2w = fc2_w + (size_t)i * CDIM * 512;
    const float* f2b = fc2_b + (size_t)i * CDIM;
    const int s = shifts[i];

    // LN1 + roll(-s) + partition -> win (buf1)
    ln_kernel<<<NROWS, 128, 0, stream>>>(t, n1g, n1b, buf1, s, 1);
    // QKV: [65536,128] x [384,128]^T -> qkvb
    gemm_wmma<<<dim3(6, 1024), 256, 0, stream>>>(
        buf1, qw, qb, nullptr, qkvb, NROWS, 128, 384, 0);
    // windowed attention -> owin
    attn_win<<<8192, 128, 0, stream>>>(qkvb, rp, owin, s);
    // proj: [65536,128] x [128,128]^T -> buf1 (window order)
    gemm_wmma<<<dim3(2, 1024), 256, 0, stream>>>(
        owin, pwm, pbm, nullptr, buf1, NROWS, 128, 128, 0);
    // reverse + roll(+s) + residual add into t
    reverse_add<<<32768, 256, 0, stream>>>(t, buf1, s);
    // LN2 -> buf1
    ln_kernel<<<NROWS, 128, 0, stream>>>(t, n2g, n2b, buf1, 0, 0);
    // fc1 + exact GELU: [65536,128] x [512,128]^T -> mbuf
    gemm_wmma<<<dim3(8, 1024), 256, 0, stream>>>(
        buf1, f1w, f1b, nullptr, mbuf, NROWS, 128, 512, 1);
    // fc2 + residual: [65536,512] x [128,512]^T + t -> t
    gemm_wmma<<<dim3(2, 1024), 256, 0, stream>>>(
        mbuf, f2w, f2b, t, t, NROWS, 512, 128, 0);
  }

  // depthwise 3x3x3 -> buf1
  dwconv<<<32768, 256, 0, stream>>>(t, dw_w, dw_b, buf1);
  // pointwise 1x1x1: [65536,128] x [256,128]^T -> zbuf
  gemm_wmma<<<dim3(4, 1024), 256, 0, stream>>>(
      buf1, pw_w, pw_b, nullptr, zbuf, NROWS, 128, 256, 0);
  // batch-norm stats + apply + ReLU + layout permute -> d_out [B,256,32^3]
  bn_stats<<<256, 256, 0, stream>>>(zbuf, stats);
  bn_apply<<<65536, 256, 0, stream>>>(zbuf, stats, bn_g, bn_b, (float*)d_out);
}
